// Attention_6983616824059
// MI455X (gfx1250) — matrix operand order repaired and checked by reading, not compiled, not run on hardware
//
#include <hip/hip_runtime.h>
#include <hip/hip_bf16.h>
#include <stdint.h>

#define B_    8
#define S_    2048
#define H_    256
#define KD_   32
#define MTILE 64      // query rows per attention workgroup
#define NKEY  64      // key rows per iteration

typedef __attribute__((ext_vector_type(16))) __bf16 v16bf;
typedef __attribute__((ext_vector_type(8)))  float  v8f;

union Frag16 {      // 16 bf16 values = 8 VGPRs, loaded as two b128s
  uint4 u[2];
  v16bf h;
};

static __device__ __forceinline__ uint16_t f32_to_bf16_bits(float f) {
  union { __bf16 h; uint16_t u; } cv;
  cv.h = (__bf16)f;
  return cv.u;
}

// ---- 16-lane butterfly reductions in pure VALU via DPP16 -------------------
// xor masks {1,2,7,15} form a GF(2) basis of the 4-bit lane space:
//   0xB1  = quad_perm [1,0,3,2]  (xor 1)
//   0x4E  = quad_perm [2,3,0,1]  (xor 2)
//   0x141 = row_half_mirror      (xor 7)
//   0x140 = row_mirror           (xor 15)
// DPP16 stays inside 16-lane rows, so lanes 0-15 and 16-31 reduce separately.
#if __has_builtin(__builtin_amdgcn_update_dpp)
#define DPP_STEP(v, ctrl, OPEXPR)                                                     \
  {                                                                                   \
    int _i = __builtin_amdgcn_update_dpp(0, __builtin_bit_cast(int, v), (ctrl),       \
                                         0xF, 0xF, true);                             \
    float _o = __builtin_bit_cast(float, _i);                                         \
    v = OPEXPR;                                                                       \
  }
static __device__ __forceinline__ float row16_max(float v) {
  DPP_STEP(v, 0xB1,  fmaxf(v, _o));
  DPP_STEP(v, 0x4E,  fmaxf(v, _o));
  DPP_STEP(v, 0x141, fmaxf(v, _o));
  DPP_STEP(v, 0x140, fmaxf(v, _o));
  return v;
}
static __device__ __forceinline__ float row16_sum(float v) {
  DPP_STEP(v, 0xB1,  (v + _o));
  DPP_STEP(v, 0x4E,  (v + _o));
  DPP_STEP(v, 0x141, (v + _o));
  DPP_STEP(v, 0x140, (v + _o));
  return v;
}
#else
static __device__ __forceinline__ float row16_max(float v) {
  for (int off = 1; off < 16; off <<= 1) v = fmaxf(v, __shfl_xor(v, off, 16));
  return v;
}
static __device__ __forceinline__ float row16_sum(float v) {
  for (int off = 1; off < 16; off <<= 1) v += __shfl_xor(v, off, 16);
  return v;
}
#endif

// =============== Phase 0a: x (f32) -> xb (bf16), grid-stride ================
__global__ void __launch_bounds__(256) cvt_x_kernel(
    const float* __restrict__ x, __hip_bfloat16* __restrict__ xb, int n) {
  for (int i = blockIdx.x * 256 + threadIdx.x; i < n; i += gridDim.x * 256)
    xb[i] = __float2bfloat16(x[i]);
}

// =============== Phase 0b: weights -> Wt[320][256] bf16 (transposed) ========
// row j: j<32 -> Wq col j ; j<64 -> Wk col j-32 ; else Wv col j-64
__global__ void __launch_bounds__(256) prep_w_kernel(
    const float* __restrict__ Wq, const float* __restrict__ Wk,
    const float* __restrict__ Wv, __hip_bfloat16* __restrict__ Wt) {
  const int j = blockIdx.x;     // 0..319 output column
  const int h = threadIdx.x;    // 0..255 contraction index
  float v;
  if (j < KD_)          v = Wq[h * KD_ + j];
  else if (j < 2 * KD_) v = Wk[h * KD_ + (j - KD_)];
  else                  v = Wv[h * H_ + (j - 2 * KD_)];
  Wt[j * H_ + h] = __float2bfloat16(v);
}

// =============== Phase 1: WMMA projection  xb[16384,256] x Wt^T -> Q,K,Vt ===
// WG = 8 waves, 64 rows x 320 cols; wave = 16-row strip x 10 col tiles.
// ALL 10 B fragments are preloaded (simultaneously live) so the register
// allocator cannot re-serialize the loop into load->wait(0)->wmma; the
// waitcnt pass must emit staggered partial waits instead.
__global__ void __launch_bounds__(256) proj_wmma_kernel(
    const uint16_t* __restrict__ xb,   // [B*S, 256] bf16
    const uint16_t* __restrict__ Wt,   // [320, 256] bf16
    const float* __restrict__ bq, const float* __restrict__ bk,
    const float* __restrict__ bv,
    __hip_bfloat16* __restrict__ Qb, __hip_bfloat16* __restrict__ Kb,
    __hip_bfloat16* __restrict__ Vt) {
  const int m0   = blockIdx.x * 64;
  const int tid  = threadIdx.x;
  const int w    = tid >> 5;
  const int lane = tid & 31;
  const int r    = w >> 1;                 // row strip
  const int cg   = w & 1;                  // col group (10 tiles of 16 = 160 cols)
  const int n    = lane & 15;
  const int hi   = (lane >= 16) ? 1 : 0;
  const int kbase = hi ? 8 : 0;            // A-layout contraction offset
  const int kb    = hi ? 8 : 0;            // B-layout contraction offset

  const v8f vzero = {};
  v8f acc[10];
  #pragma unroll
  for (int i = 0; i < 10; ++i) acc[i] = vzero;

  const int arow = m0 + r * 16 + n;        // A row handled by this lane

  #pragma unroll 1
  for (int kt = 0; kt < H_; kt += 32) {
    Frag16 a;
    const uint16_t* ap = xb + (size_t)arow * H_ + kt + kbase;
    a.u[0] = *(const uint4*)(ap);
    a.u[1] = *(const uint4*)(ap + 16);

    const uint16_t* wbase = Wt + (size_t)(cg * 160 + n) * H_ + kt + kb;
    Frag16 bw[10];
    #pragma unroll
    for (int c = 0; c < 10; ++c) {
      const uint16_t* wp = wbase + (size_t)c * 16 * H_;
      bw[c].u[0] = *(const uint4*)(wp);
      bw[c].u[1] = *(const uint4*)(wp + 16);
    }
    #pragma unroll
    for (int c = 0; c < 10; ++c)
      acc[c] = __builtin_amdgcn_wmma_f32_16x16x32_bf16(false, a.h, false, bw[c].h,
                                                       (short)0, acc[c], false, false);
  }

  const int b = m0 / S_;                   // 64-row block never straddles a batch
  #pragma unroll
  for (int c = 0; c < 10; ++c) {
    const int col = cg * 160 + c * 16 + n;
    const float bias = (col < KD_) ? bq[col]
                     : (col < 2 * KD_) ? bk[col - KD_] : bv[col - 2 * KD_];
    #pragma unroll
    for (int rr = 0; rr < 8; ++rr) {
      const int row = m0 + r * 16 + rr + hi * 8;    // global row = b*S + s
      const int s   = row - b * S_;
      const __hip_bfloat16 hv = __float2bfloat16(acc[c][rr] + bias);
      if (col < KD_)          Qb[(size_t)row * KD_ + col] = hv;
      else if (col < 2 * KD_) Kb[(size_t)row * KD_ + (col - KD_)] = hv;
      else                    Vt[((size_t)b * H_ + (col - 2 * KD_)) * S_ + s] = hv;
    }
  }
}

// =============== Phase 2: flash attention, bf16 WMMA, f32 softmax/accum =====
__global__ void __launch_bounds__(256) attn_kernel(
    const float*    __restrict__ x,
    const float*    __restrict__ gammap,
    const uint16_t* __restrict__ Qb,
    const uint16_t* __restrict__ Kb,
    const uint16_t* __restrict__ Vt,
    float*          __restrict__ out) {
  __shared__ uint16_t Plds[8][16][NKEY];   // per-wave 16x64 bf16 probabilities (16 KB)

  const int blk  = blockIdx.x;
  const int b    = blk / (S_ / MTILE);
  const int m0   = (blk % (S_ / MTILE)) * MTILE;
  const int tid  = threadIdx.x;
  const int w    = tid >> 5;
  const int lane = tid & 31;
  const int r    = w >> 1;       // row strip
  const int ch   = w & 1;        // output column half
  const int n    = lane & 15;
  const int hi   = (lane >= 16) ? 1 : 0;
  const int kbase = hi ? 8 : 0;
  const int kb    = hi ? 8 : 0;
  const size_t bS = (size_t)b * S_;

  Frag16 aq;
  {
    const uint16_t* qp = Qb + (bS + m0 + r * 16 + n) * KD_ + kbase;
    aq.u[0] = *(const uint4*)(qp);
    aq.u[1] = *(const uint4*)(qp + 16);
  }

  const v8f vzero = {};
  v8f acc[8];
  #pragma unroll
  for (int i = 0; i < 8; ++i) acc[i] = vzero;
  float rmax[8], rsum[8];
  #pragma unroll
  for (int i = 0; i < 8; ++i) { rmax[i] = -3.0e38f; rsum[i] = 0.0f; }

  // V fragments: two groups of 4 (each covers 2 col tiles x 2 key subtiles).
  Frag16 vf[2][4];
  const uint16_t* vwave = Vt + ((size_t)b * H_ + ch * 128 + n) * S_ + kb;
  auto loadV = [&](Frag16* f, int cp, int kt) {
    const uint16_t* vp0 = vwave + (size_t)(2 * cp) * 16 * S_ + kt;   // col tile 2cp
    const uint16_t* vp1 = vp0 + (size_t)16 * S_;                     // col tile 2cp+1
    f[0].u[0] = *(const uint4*)(vp0);      f[0].u[1] = *(const uint4*)(vp0 + 16);
    f[1].u[0] = *(const uint4*)(vp0 + 32); f[1].u[1] = *(const uint4*)(vp0 + 48);
    f[2].u[0] = *(const uint4*)(vp1);      f[2].u[1] = *(const uint4*)(vp1 + 16);
    f[3].u[0] = *(const uint4*)(vp1 + 32); f[3].u[1] = *(const uint4*)(vp1 + 48);
  };

  #pragma unroll 1
  for (int kt = 0; kt < S_; kt += NKEY) {
    // ---- scores: load all 4 K B-fragments (one clause), then 4 WMMAs
    Frag16 bkf[4];
    #pragma unroll
    for (int t = 0; t < 4; ++t) {
      const uint16_t* kp = Kb + (bS + kt + t * 16 + n) * KD_ + kb;
      bkf[t].u[0] = *(const uint4*)(kp);
      bkf[t].u[1] = *(const uint4*)(kp + 16);
    }
    v8f sc[4];
    #pragma unroll
    for (int t = 0; t < 4; ++t)
      sc[t] = __builtin_amdgcn_wmma_f32_16x16x32_bf16(false, aq.h, false, bkf[t].h,
                                                      (short)0, vzero, false, false);

    // Issue BOTH V groups' loads now (16 b128s): both stay live through the
    // softmax below, so their latency is hidden under ~150 VALU ops + the
    // LDS round-trip, and the allocator cannot collapse them.
    loadV(vf[0], 0, kt);
    loadV(vf[1], 1, kt);

    // ---- online softmax; all cross-lane work done with DPP16 (no DS traffic)
    float nmax[8], scl[8];
    #pragma unroll
    for (int rr = 0; rr < 8; ++rr) {
      float m = fmaxf(fmaxf(sc[0][rr], sc[1][rr]), fmaxf(sc[2][rr], sc[3][rr]));
      m = row16_max(m);
      nmax[rr] = fmaxf(rmax[rr], m);
      scl[rr]  = __expf(rmax[rr] - nmax[rr]);
    }
    float tsum[8];
    #pragma unroll
    for (int rr = 0; rr < 8; ++rr) tsum[rr] = 0.0f;
    #pragma unroll
    for (int t = 0; t < 4; ++t) {
      #pragma unroll
      for (int rr = 0; rr < 8; ++rr) {
        float p = __expf(sc[t][rr] - nmax[rr]);
        tsum[rr] += p;
        Plds[w][rr + hi * 8][t * 16 + n] = f32_to_bf16_bits(p);
      }
    }
    #pragma unroll
    for (int rr = 0; rr < 8; ++rr) {
      rsum[rr] = rsum[rr] * scl[rr] + row16_sum(tsum[rr]);
      rmax[rr] = nmax[rr];
    }
    #pragma unroll
    for (int ct = 0; ct < 8; ++ct) {
      #pragma unroll
      for (int rr = 0; rr < 8; ++rr) acc[ct][rr] *= scl[rr];
    }

    // LDS region is wave-private: only store->load ordering within this wave
    asm volatile("s_wait_dscnt 0" ::: "memory");

    // ---- reload P as two A-layout fragments (keys kt..kt+63)
    Frag16 pa0, pa1;
    {
      const uint16_t* pp = &Plds[w][n][0];
      pa0.u[0] = *(const uint4*)(pp + kbase);
      pa0.u[1] = *(const uint4*)(pp + kbase + 16);
      pa1.u[0] = *(const uint4*)(pp + kbase + 32);
      pa1.u[1] = *(const uint4*)(pp + kbase + 48);
    }

    // ---- P x V : consume group cp, then reload that buffer for cp+2.
    // The WAR dependence on the just-consumed registers pins each reload
    // after its consumer WMMAs, keeping a structural distance-2 pipeline.
    #pragma unroll
    for (int cp = 0; cp < 4; ++cp) {
      const Frag16* f = vf[cp & 1];
      const int ct0 = 2 * cp, ct1 = 2 * cp + 1;
      acc[ct0] = __builtin_amdgcn_wmma_f32_16x16x32_bf16(false, pa0.h, false, f[0].h,
                                                         (short)0, acc[ct0], false, false);
      acc[ct0] = __builtin_amdgcn_wmma_f32_16x16x32_bf16(false, pa1.h, false, f[1].h,
                                                         (short)0, acc[ct0], false, false);
      acc[ct1] = __builtin_amdgcn_wmma_f32_16x16x32_bf16(false, pa0.h, false, f[2].h,
                                                         (short)0, acc[ct1], false, false);
      acc[ct1] = __builtin_amdgcn_wmma_f32_16x16x32_bf16(false, pa1.h, false, f[3].h,
                                                         (short)0, acc[ct1], false, false);
      if (cp + 2 < 4) loadV(vf[cp & 1], cp + 2, kt);
    }
  }

  // ---- epilogue: normalize, gamma * out + x
  const float gamma = gammap[0];
  float ginv[8];
  #pragma unroll
  for (int rr = 0; rr < 8; ++rr) ginv[rr] = gamma / rsum[rr];
  #pragma unroll
  for (int ct = 0; ct < 8; ++ct) {
    const int col = ch * 128 + ct * 16 + n;
    #pragma unroll
    for (int rr = 0; rr < 8; ++rr) {
      const int row = m0 + r * 16 + rr + hi * 8;
      const size_t idx = (bS + row) * H_ + col;
      out[idx] = acc[ct][rr] * ginv[rr] + x[idx];
    }
  }
}

// ---------------- launch ----------------
extern "C" void kernel_launch(void* const* d_in, const int* in_sizes, int n_in,
                              void* d_out, int out_size, void* d_ws, size_t ws_size,
                              hipStream_t stream) {
  const float* x     = (const float*)d_in[0];
  const float* Wq    = (const float*)d_in[1];
  const float* bq    = (const float*)d_in[2];
  const float* Wk    = (const float*)d_in[3];
  const float* bk    = (const float*)d_in[4];
  const float* Wv    = (const float*)d_in[5];
  const float* bv    = (const float*)d_in[6];
  const float* gamma = (const float*)d_in[7];

  char* ws = (char*)d_ws;
  const size_t qk_bytes = (size_t)B_ * S_ * KD_ * 2;       // 1 MB each
  const size_t vt_bytes = (size_t)B_ * H_  * S_  * 2;      // 8 MB
  const size_t xb_bytes = (size_t)B_ * S_ * H_ * 2;        // 8 MB
  __hip_bfloat16* Qb = (__hip_bfloat16*)(ws);
  __hip_bfloat16* Kb = (__hip_bfloat16*)(ws + qk_bytes);
  __hip_bfloat16* Vt = (__hip_bfloat16*)(ws + 2 * qk_bytes);
  __hip_bfloat16* xb = (__hip_bfloat16*)(ws + 2 * qk_bytes + vt_bytes);
  __hip_bfloat16* Wt = (__hip_bfloat16*)(ws + 2 * qk_bytes + vt_bytes + xb_bytes);

  cvt_x_kernel<<<4096, 256, 0, stream>>>(x, xb, B_ * S_ * H_);
  prep_w_kernel<<<2 * KD_ + H_, 256, 0, stream>>>(Wq, Wk, Wv, Wt);
  proj_wmma_kernel<<<(B_ * S_) / 64, 256, 0, stream>>>(
      (const uint16_t*)xb, (const uint16_t*)Wt, bq, bk, bv, Qb, Kb, Vt);
  attn_kernel<<<B_ * (S_ / MTILE), 256, 0, stream>>>(
      x, gamma, (const uint16_t*)Qb, (const uint16_t*)Kb, (const uint16_t*)Vt,
      (float*)d_out);
}
